// DND_8744553415037
// MI455X (gfx1250) — matrix operand order repaired and checked by reading, not compiled, not run on hardware
//
#include <hip/hip_runtime.h>
#include <hip/hip_bf16.h>
#include <stdint.h>

#define D       128
#define TK      16      // keys per WMMA tile (N)
#define NWAVES  8
#define WGQ     16      // queries per workgroup (M)
#define NB      512     // radix histogram bins (9 bits max per pass)
#define KMAX    64      // upper bound on num_neighbours used for ws sizing
#define BIAS    1024.0f // makes shifted distances strictly positive (Cauchy-Schwarz bound)
#define SCALE   1048576.0f  // 2^20 fixed-point scale: key = (dist+BIAS)*SCALE < 2^31

typedef __attribute__((ext_vector_type(16))) __bf16 v16bf;
typedef __attribute__((ext_vector_type(8)))  float  v8f;

union BF16x16 { v16bf v; uint32_t w[8]; };

__device__ __forceinline__ uint32_t pack_bf16_trunc(float lo, float hi) {
  return (__float_as_uint(hi) & 0xFFFF0000u) | (__float_as_uint(lo) >> 16);
}
__device__ __forceinline__ uint16_t f2bf_rne(float f) {
  uint32_t u = __float_as_uint(f);
  return (uint16_t)((u + 0x7FFFu + ((u >> 16) & 1u)) >> 16);
}
__device__ __forceinline__ uint32_t pack_bf16_rne(float lo, float hi) {
  return (uint32_t)f2bf_rne(lo) | ((uint32_t)f2bf_rne(hi) << 16);
}

// ---------------- Kernel 1: per-key squared norms ----------------
__global__ __launch_bounds__(256) void ksq_kernel(const float* __restrict__ keys,
                                                  float* __restrict__ ksq, int cap) {
  int i = blockIdx.x * blockDim.x + threadIdx.x;
  if (i >= cap) return;
  const float4* row = (const float4*)(keys + (size_t)i * D);
  float s = 0.f;
#pragma unroll
  for (int j = 0; j < D / 4; ++j) {
    float4 v = row[j];
    s += v.x * v.x + v.y * v.y + v.z * v.z + v.w * v.w;
  }
  ksq[i] = s;
}

// ---------------- Kernel 1b: zero the neighbour index buffer ------
__global__ void nbr_init_kernel(int* __restrict__ nbr, const int* __restrict__ kptr, int B) {
  int k = *kptr;
  int i = blockIdx.x * blockDim.x + threadIdx.x;
  if (i < B * k) nbr[i] = 0;
}

// ---------------- Kernel 1c: pre-pack keys as bf16 WMMA B-fragments ----------
// dwords [ ((t*4 + c)*32 + L)*8 .. +8 ) = elements e=0..15 of lane L's v16bf,
// i.e. keys[t*16 + (L%16)][c*32 + 16*(L/16) + e]; the wave's fragment loads are
// fully coalesced (consecutive lanes -> consecutive 32B blocks).
__global__ __launch_bounds__(256) void pack_keys_kernel(const float* __restrict__ keys,
                                                        uint32_t* __restrict__ pk, int cap) {
  int gid = blockIdx.x * 256 + threadIdx.x;
  int total = (cap / TK) * 4 * 32;
  if (gid >= total) return;
  int L = gid & 31;
  int c = (gid >> 5) & 3;
  int t = gid >> 7;
  int row  = t * TK + (L & 15);
  int col0 = c * 32 + (L >> 4) * 16;
  const float4* src = (const float4*)(keys + (size_t)row * D + col0);
  uint32_t w[8];
#pragma unroll
  for (int j = 0; j < 4; ++j) {
    float4 v = src[j];
    w[2 * j]     = pack_bf16_rne(v.x, v.y);
    w[2 * j + 1] = pack_bf16_rne(v.z, v.w);
  }
  uint4* dst = (uint4*)(pk + (size_t)gid * 8);
  dst[0] = make_uint4(w[0], w[1], w[2], w[3]);
  dst[1] = make_uint4(w[4], w[5], w[6], w[7]);
}

// ---------------- Kernel 2: WMMA distance GEMM + exact radix top-K ------------
template <bool PRE>
__global__ __launch_bounds__(256) void knn_select_kernel(
    const float* __restrict__ queries, const float* __restrict__ keys,
    const uint32_t* __restrict__ pkeys, const float* __restrict__ ksqbuf,
    const int* __restrict__ kptr, int* __restrict__ nbr, int cap) {
  __shared__ uint32_t hist[WGQ * NB];        // 32 KB
  __shared__ uint16_t qs[WGQ * D];           // 4 KB query tile (bf16 bits)
  __shared__ uint32_t prefix[WGQ], remain[WGQ], cl[WGQ], ce[WGQ];

  const int k    = *kptr;
  const int tid  = threadIdx.x;
  const int lane = tid & 31;
  const int wave = tid >> 5;
  const int qb   = blockIdx.x * WGQ;
  const int n    = lane & 15;    // C-layout: column of the 16x16 tile
  const int half = lane >> 4;    // C-layout: lane group (M offset 0 or 8)
  const int ntiles = cap / TK;

  // Stage query tile as bf16 rows in LDS.
  {
    int m = tid >> 4, g = tid & 15;
    const float* qrow = queries + (size_t)(qb + m) * D + g * 8;
#pragma unroll
    for (int j = 0; j < 8; ++j)
      qs[m * D + g * 8 + j] = f2bf_rne(qrow[j]);
  }
  if (tid < WGQ) { prefix[tid] = 0u; remain[tid] = (uint32_t)k; cl[tid] = 0u; ce[tid] = 0u; }
  __syncthreads();

  // A fragments (16x32 bf16 per K-chunk), contiguous k order (same as B fragments):
  // lane L (m = L%16, half = L/16): elems 0..15 = cols 32c + 16*half + 0..15
  BF16x16 afrag[4];
  {
    int m = n;
#pragma unroll
    for (int c = 0; c < 4; ++c) {
      const uint32_t* p0 = (const uint32_t*)&qs[m * D + c * 32 + half * 16];
      const uint32_t* p1 = (const uint32_t*)&qs[m * D + c * 32 + half * 16 + 8];
#pragma unroll
      for (int j = 0; j < 4; ++j) { afrag[c].w[j] = p0[j]; afrag[c].w[4 + j] = p1[j]; }
    }
  }

  // Radix key: u32((dist + BIAS) * 2^20), a monotone fixed-point map. 31 bits.
  // Passes 0..3 select on 9/9/9/4-bit digits; pass 4 collects exactly k per query.
  // (Integer key keeps pass-0 digits spread over many bins -> no LDS atomic
  //  hot-spotting, unlike float-bit keys whose top digit is exponent-only.)
  for (int p = 0; p < 5; ++p) {
    const bool collect = (p == 4);
    const int bits  = (p == 3) ? 4 : 9;
    const int shift = (p == 0) ? 22 : (p == 1) ? 13 : (p == 2) ? 4 : 0;
    const int nbins = 1 << bits;

    if (!collect)
      for (int i = tid; i < WGQ * NB; i += 256) hist[i] = 0u;
    __syncthreads();

    uint32_t pm[8], kth[8], Rm[8], eb[8];
    int* nbq[8];
#pragma unroll
    for (int r = 0; r < 8; ++r) {
      int m = r + 8 * half;
      pm[r]  = prefix[m];
      kth[r] = prefix[m];
      Rm[r]  = remain[m];
      eb[r]  = (uint32_t)k - Rm[r];                 // slot base for equal-key entries
      nbq[r] = nbr + (size_t)(qb + m) * k;          // per-row output base
    }

    // Compute one key tile: fragment loads + 4-chunk WMMA accumulation.
    auto compute_tile = [&](int t, v8f& accO, float& ksqO, int& krO) {
      const int kr = t * TK + n;
      const float ksqs = (ksqbuf[kr] + BIAS) * SCALE;
      v8f acc = {};
#pragma unroll
      for (int c = 0; c < 4; ++c) {
        BF16x16 bfrag;
        if (PRE) {
          const uint4* p4 = (const uint4*)(pkeys + ((((size_t)t * 4 + c) * 32) + lane) * 8);
          uint4 a0 = p4[0], a1 = p4[1];
          bfrag.w[0] = a0.x; bfrag.w[1] = a0.y; bfrag.w[2] = a0.z; bfrag.w[3] = a0.w;
          bfrag.w[4] = a1.x; bfrag.w[5] = a1.y; bfrag.w[6] = a1.z; bfrag.w[7] = a1.w;
        } else {
          const float4* kp = (const float4*)(keys + (size_t)kr * D + c * 32 + half * 16);
#pragma unroll
          for (int j = 0; j < 4; ++j) {
            float4 v = kp[j];
            bfrag.w[2 * j]     = pack_bf16_trunc(v.x, v.y);
            bfrag.w[2 * j + 1] = pack_bf16_trunc(v.z, v.w);
          }
        }
        acc = __builtin_amdgcn_wmma_f32_16x16x32_bf16(
            false, afrag[c].v, false, bfrag.v, (short)0, acc, false, false);
      }
      accO = acc; ksqO = ksqs; krO = kr;
    };

    // Distance -> fixed-point radix key -> histogram / collect for one tile.
    auto epilogue = [&](v8f acc, float ksqs, int kr) {
#pragma unroll
      for (int r = 0; r < 8; ++r) {
        const int m = r + 8 * half;   // query row of this accumulator
        const uint32_t key = (uint32_t)fmaf(-2.0f * SCALE, acc[r], ksqs);
        if (!collect) {
          bool match = true;
          if (p > 0) match = ((key >> (shift + bits)) == pm[r]);
          if (match) atomicAdd(&hist[m * NB + ((key >> shift) & (nbins - 1))], 1u);
        } else {
          if (key < kth[r]) {
            uint32_t pos = atomicAdd(&cl[m], 1u);
            if ((int)pos < k) nbq[r][pos] = kr;
          } else if (key == kth[r]) {
            uint32_t e = atomicAdd(&ce[m], 1u);
            if (e < Rm[r]) nbq[r][eb[r] + e] = kr;
          }
        }
      }
    };

    // Software pipeline: epilogue of tile t runs in the shadow of tile t+1's
    // WMMA chain (fills the WMMA->VALU hazard slots with useful work).
    v8f pacc; float pksq; int pkr;
    compute_tile(wave, pacc, pksq, pkr);
    for (int t = wave + NWAVES; t < ntiles; t += NWAVES) {
      v8f acc; float kq; int kr2;
      compute_tile(t, acc, kq, kr2);
      epilogue(pacc, pksq, pkr);
      pacc = acc; pksq = kq; pkr = kr2;
    }
    epilogue(pacc, pksq, pkr);
    __syncthreads();

    if (!collect && tid < WGQ) {
      uint32_t rem = remain[tid], cum = 0u; int sel = nbins - 1;
      for (int b = 0; b < nbins; ++b) {
        uint32_t c = hist[tid * NB + b];
        if (cum + c >= rem) { sel = b; rem -= cum; break; }
        cum += c;
      }
      remain[tid] = rem;                                   // rank within equal-key group
      prefix[tid] = (prefix[tid] << bits) | (uint32_t)sel; // after pass 3: exact kth key
    }
    __syncthreads();
  }
}

// ---------------- Kernel 3: exact fp32 weights + reduction (1 wave / query) ----
__global__ __launch_bounds__(256) void finalize_kernel(
    const float* __restrict__ queries, const float* __restrict__ keys,
    const float* __restrict__ values, const int* __restrict__ kptr,
    const int* __restrict__ nbr, float* __restrict__ out, int cap) {
  const int k    = *kptr;
  const int lane = threadIdx.x & 31;
  const int wave = threadIdx.x >> 5;
  const int b    = blockIdx.x * NWAVES + wave;
  const float4 qv = ((const float4*)(queries + (size_t)b * D))[lane];
  float wsum = 0.f, vsum = 0.f;
  for (int j = 0; j < k; ++j) {
    int idx = nbr[(size_t)b * k + j];
    if (idx < 0 || idx >= cap) idx = 0;  // defensive
    float4 kv = ((const float4*)(keys + (size_t)idx * D))[lane];
    float dx = qv.x - kv.x, dy = qv.y - kv.y, dz = qv.z - kv.z, dw = qv.w - kv.w;
    float s = dx * dx + dy * dy + dz * dz + dw * dw;
#pragma unroll
    for (int o = 16; o >= 1; o >>= 1) s += __shfl_xor(s, o, 32);
    float w = 1.0f / (sqrtf(s + 1e-8f) + 1e-3f);
    wsum += w;
    vsum += w * values[idx];
  }
  if (lane == 0) out[b] = vsum / wsum;
}

extern "C" void kernel_launch(void* const* d_in, const int* in_sizes, int n_in,
                              void* d_out, int out_size, void* d_ws, size_t ws_size,
                              hipStream_t stream) {
  const float* queries = (const float*)d_in[0];
  const float* keys    = (const float*)d_in[1];
  const float* values  = (const float*)d_in[2];
  const int*   kptr    = (const int*)d_in[3];

  const int B   = in_sizes[0] / D;   // 2048
  const int cap = in_sizes[1] / D;   // 131072

  // Workspace layout: [ksq: cap f32][nbr: B*KMAX i32][packed bf16 keys: cap*D bf16]
  const size_t off_nbr = (size_t)cap * sizeof(float);
  const size_t off_pk  = off_nbr + (size_t)B * KMAX * sizeof(int);
  const size_t need_pk = off_pk + (size_t)cap * D * sizeof(uint16_t);

  float*    ksq = (float*)d_ws;
  int*      nbr = (int*)((char*)d_ws + off_nbr);
  uint32_t* pk  = (uint32_t*)((char*)d_ws + off_pk);
  const bool pre = (ws_size >= need_pk);

  ksq_kernel<<<(cap + 255) / 256, 256, 0, stream>>>(keys, ksq, cap);
  nbr_init_kernel<<<(B * KMAX + 255) / 256, 256, 0, stream>>>(nbr, kptr, B);
  if (pre) {
    const int total = (cap / TK) * 4 * 32;
    pack_keys_kernel<<<(total + 255) / 256, 256, 0, stream>>>(keys, pk, cap);
    knn_select_kernel<true><<<B / WGQ, 256, 0, stream>>>(queries, keys, pk, ksq, kptr, nbr, cap);
  } else {
    knn_select_kernel<false><<<B / WGQ, 256, 0, stream>>>(queries, keys, pk, ksq, kptr, nbr, cap);
  }
  finalize_kernel<<<B / NWAVES, 256, 0, stream>>>(queries, keys, values, kptr, nbr,
                                                  (float*)d_out, cap);
}
